// CMFA_6021544149280
// MI455X (gfx1250) — matrix operand order repaired and checked by reading, not compiled, not run on hardware
//
#include <hip/hip_runtime.h>
#include <stdint.h>

typedef __attribute__((ext_vector_type(16))) __bf16 v16bf;
typedef __attribute__((ext_vector_type(8)))  __bf16 v8bf;
typedef __attribute__((ext_vector_type(4)))  __bf16 v4bf;
typedef __attribute__((ext_vector_type(8)))  float  v8f;
typedef int v4i_vs __attribute__((vector_size(16)));   // matches builtin proto 'V4i'

#define BATCH  32768
#define HIDDEN 512
#define IMG    2048
#define TAB    128
#define BM 128
#define BN 128
#define BK 32
#define LDSS 40   // bf16 elems per LDS row: 80B stride = 16B-aligned, 20-bank step

#if __has_builtin(__builtin_amdgcn_global_load_async_to_lds_b128)
#define USE_ASYNC 1
#define GV(p) ((__attribute__((address_space(1))) v4i_vs*)(p))
#define LV(p) ((__attribute__((address_space(3))) v4i_vs*)(p))
#if __has_builtin(__builtin_amdgcn_s_wait_asynccnt)
#define WAIT_ASYNC(n) __builtin_amdgcn_s_wait_asynccnt(n)
#else
#define WAIT_ASYNC(n) asm volatile("s_wait_asynccnt %0" :: "i"(n) : "memory")
#endif
#else
#define USE_ASYNC 0
#endif

static __device__ __forceinline__ __bf16 f2bf(float f) {
    // round-to-nearest-even fp32 -> bf16
    uint32_t u = __builtin_bit_cast(uint32_t, f);
    uint32_t r = u + 0x7FFFu + ((u >> 16) & 1u);
    uint16_t h = (uint16_t)(r >> 16);
    return __builtin_bit_cast(__bf16, h);
}
static __device__ __forceinline__ float bf2f(__bf16 b) {
    uint32_t u = ((uint32_t)__builtin_bit_cast(uint16_t, b)) << 16;
    return __builtin_bit_cast(float, u);
}

// elementwise fp32 -> bf16 (n must be a multiple of 4)
__global__ __launch_bounds__(256)
void cvt_f32_bf16_kernel(const float* __restrict__ s, __bf16* __restrict__ d,
                         long long n4) {
    long long i = (long long)blockIdx.x * blockDim.x + threadIdx.x;
    if (i < n4) {
        float4 v = ((const float4*)s)[i];
        v4bf b = { f2bf(v.x), f2bf(v.y), f2bf(v.z), f2bf(v.w) };
        ((v4bf*)d)[i] = b;
    }
}

// C[m,n] = act( sum_k A[m,k]*W[n,k] + bias[n] (+ resid[m,n]) )
// A: [BATCH,K] bf16 row-major; W: [HIDDEN,K] bf16 row-major (torch Linear layout)
// OUT_BF16: C is __bf16* [BATCH,HIDDEN]; else float* with leading dim ldc.
template <int DO_RELU, int OUT_BF16, int HAS_RESID>
__global__ __launch_bounds__(256)
void gemm_bf16_wmma_kernel(const __bf16* __restrict__ A,
                           const __bf16* __restrict__ W,
                           const float*  __restrict__ bias,
                           const __bf16* __restrict__ resid,
                           void* __restrict__ Cout,
                           int K, int ldc)
{
    __shared__ __align__(16) __bf16 As[2][BM * LDSS];
    __shared__ __align__(16) __bf16 Ws[2][BN * LDSS];

    const int tid  = threadIdx.x;
    const int lane = tid & 31;
    const int wave = tid >> 5;
    const int wr   = wave & 3;    // 4 row groups x 32 rows
    const int wc   = wave >> 2;   // 2 col groups x 64 cols
    const int hl   = lane >> 4;   // half-wave select (ISA 16-bit operand layout)
    const int l16  = lane & 15;

    const int rowBase = blockIdx.y * BM;
    const int colBase = blockIdx.x * BN;

    // Per-thread staging coordinates: 512 16B chunks per tile, 2 per thread.
    const int srow0 = tid >> 2;          // chunk row, j=0
    const int srow1 = (tid + 256) >> 2;  // chunk row, j=1
    const int sseg  = tid & 3;           // 16B chunk within a 64B row

    auto stage = [&](int buf, int k0) {
#if USE_ASYNC
        __builtin_amdgcn_global_load_async_to_lds_b128(
            GV(A + (size_t)(rowBase + srow0) * K + k0 + sseg * 8),
            LV(&As[buf][srow0 * LDSS + sseg * 8]), 0, 0);
        __builtin_amdgcn_global_load_async_to_lds_b128(
            GV(W + (size_t)(colBase + srow0) * K + k0 + sseg * 8),
            LV(&Ws[buf][srow0 * LDSS + sseg * 8]), 0, 0);
        __builtin_amdgcn_global_load_async_to_lds_b128(
            GV(A + (size_t)(rowBase + srow1) * K + k0 + sseg * 8),
            LV(&As[buf][srow1 * LDSS + sseg * 8]), 0, 0);
        __builtin_amdgcn_global_load_async_to_lds_b128(
            GV(W + (size_t)(colBase + srow1) * K + k0 + sseg * 8),
            LV(&Ws[buf][srow1 * LDSS + sseg * 8]), 0, 0);
#else
        *(v8bf*)&As[buf][srow0 * LDSS + sseg * 8] =
            *(const v8bf*)(A + (size_t)(rowBase + srow0) * K + k0 + sseg * 8);
        *(v8bf*)&Ws[buf][srow0 * LDSS + sseg * 8] =
            *(const v8bf*)(W + (size_t)(colBase + srow0) * K + k0 + sseg * 8);
        *(v8bf*)&As[buf][srow1 * LDSS + sseg * 8] =
            *(const v8bf*)(A + (size_t)(rowBase + srow1) * K + k0 + sseg * 8);
        *(v8bf*)&Ws[buf][srow1 * LDSS + sseg * 8] =
            *(const v8bf*)(W + (size_t)(colBase + srow1) * K + k0 + sseg * 8);
#endif
    };

    v8f acc[2][4];
    #pragma unroll
    for (int i = 0; i < 2; ++i)
        #pragma unroll
        for (int j = 0; j < 4; ++j)
            #pragma unroll
            for (int r = 0; r < 8; ++r)
                acc[i][j][r] = 0.0f;

    stage(0, 0);   // prime buffer 0 (4 async loads outstanding per wave)

    const int nTiles = K / BK;
    for (int t = 0; t < nTiles; ++t) {
        const int cur = t & 1;
        __syncthreads();   // everyone is done reading the buffer we stage next
        if (t + 1 < nTiles) {
            stage(cur ^ 1, (t + 1) * BK);  // prefetch next tile into other buffer
#if USE_ASYNC
            WAIT_ASYNC(4);                 // cur's 4 loads done; next's may fly
#endif
        } else {
#if USE_ASYNC
            WAIT_ASYNC(0);
#endif
        }
        __syncthreads();   // staged data visible block-wide

        // B fragments (32x16 bf16): lane (hl,n) holds W[n][hl*16 .. hl*16+15]
        v16bf bfrag[4];
        #pragma unroll
        for (int j = 0; j < 4; ++j) {
            const int n = wc * 64 + j * 16 + l16;
            const v8bf b0 = *(const v8bf*)&Ws[cur][n * LDSS + hl * 16];
            const v8bf b1 = *(const v8bf*)&Ws[cur][n * LDSS + hl * 16 + 8];
            bfrag[j] = __builtin_shufflevector(b0, b1, 0,1,2,3,4,5,6,7,8,9,10,11,12,13,14,15);
        }
        // A fragments (16x32 bf16): lane (hl,m) holds A[m][hl*16 .. hl*16+15]
        #pragma unroll
        for (int i = 0; i < 2; ++i) {
            const int m = wr * 32 + i * 16 + l16;
            const v8bf a0 = *(const v8bf*)&As[cur][m * LDSS + hl * 16];
            const v8bf a1 = *(const v8bf*)&As[cur][m * LDSS + hl * 16 + 8];
            const v16bf afrag = __builtin_shufflevector(a0, a1, 0,1,2,3,4,5,6,7,8,9,10,11,12,13,14,15);
            #pragma unroll
            for (int j = 0; j < 4; ++j) {
                acc[i][j] = __builtin_amdgcn_wmma_f32_16x16x32_bf16(
                    false, afrag, false, bfrag[j], (short)0, acc[i][j], false, false);
            }
        }
    }

    // Epilogue (branch-free via template params):
    // C/D layout — VGPR r holds row r + 8*hl, column = lane%16.
    #pragma unroll
    for (int i = 0; i < 2; ++i) {
        const int mrow0 = rowBase + wr * 32 + i * 16 + hl * 8;
        #pragma unroll
        for (int j = 0; j < 4; ++j) {
            const int col = colBase + wc * 64 + j * 16 + l16;
            const float b = bias[col];
            #pragma unroll
            for (int r = 0; r < 8; ++r) {
                const int mrow = mrow0 + r;
                float v = acc[i][j][r] + b;
                if (HAS_RESID) v += bf2f(resid[(size_t)mrow * HIDDEN + col]);
                if (DO_RELU)   v = fmaxf(v, 0.0f);
                if (OUT_BF16)
                    ((__bf16*)Cout)[(size_t)mrow * HIDDEN + col] = f2bf(v);
                else
                    ((float*)Cout)[(size_t)mrow * ldc + col] = v;
            }
        }
    }
}

extern "C" void kernel_launch(void* const* d_in, const int* in_sizes, int n_in,
                              void* d_out, int out_size, void* d_ws, size_t ws_size,
                              hipStream_t stream)
{
    // setup_inputs() dict order
    const float* i_in  = (const float*)d_in[0];
    const float* t_in  = (const float*)d_in[1];
    const float* fi1_w = (const float*)d_in[2];
    const float* fi1_b = (const float*)d_in[3];
    const float* fi2_w = (const float*)d_in[4];
    const float* fi2_b = (const float*)d_in[5];
    const float* ft1_w = (const float*)d_in[6];
    const float* ft1_b = (const float*)d_in[7];
    const float* ft2_w = (const float*)d_in[8];
    const float* ft2_b = (const float*)d_in[9];
    const float* ci1_w = (const float*)d_in[10];
    const float* ci1_b = (const float*)d_in[11];
    // ci2/ci3/ct2/ct3 and all q/k projections: softmax over a single key == 1.0
    // exactly, so they are forward-exact no-ops; skipped.
    const float* ct1_w = (const float*)d_in[16];
    const float* ct1_b = (const float*)d_in[17];
    const float* aV_wv = (const float*)d_in[26];
    const float* aV_bv = (const float*)d_in[27];
    const float* aV_wo = (const float*)d_in[28];
    const float* aV_bo = (const float*)d_in[29];
    const float* aT_wv = (const float*)d_in[34];
    const float* aT_bv = (const float*)d_in[35];
    const float* aT_wo = (const float*)d_in[36];
    const float* aT_bo = (const float*)d_in[37];

    float* out = (float*)d_out;   // [BATCH, 1024] = concat(V_, T_)

    // Workspace layout (all bf16, 256B-aligned slices)
    char* wsp = (char*)d_ws;
    auto take = [&](size_t elems) {
        __bf16* p = (__bf16*)wsp;
        wsp += (elems * sizeof(__bf16) + 255) & ~(size_t)255;
        return p;
    };
    __bf16* iB   = take((size_t)BATCH * IMG);      // 128 MB
    __bf16* tB   = take((size_t)BATCH * TAB);      //   8 MB
    __bf16* a0   = take((size_t)BATCH * HIDDEN);   //  32 MB  (i_)
    __bf16* a1   = take((size_t)BATCH * HIDDEN);   //  32 MB  (t_)
    __bf16* a2   = take((size_t)BATCH * HIDDEN);   //  32 MB
    __bf16* a3   = take((size_t)BATCH * HIDDEN);   //  32 MB
    __bf16* wfi1 = take((size_t)HIDDEN * IMG);
    __bf16* wft1 = take((size_t)HIDDEN * TAB);
    __bf16* wci1 = take((size_t)HIDDEN * HIDDEN);
    __bf16* wct1 = take((size_t)HIDDEN * HIDDEN);
    __bf16* wavv = take((size_t)HIDDEN * HIDDEN);
    __bf16* wavo = take((size_t)HIDDEN * HIDDEN);
    __bf16* watv = take((size_t)HIDDEN * HIDDEN);
    __bf16* wato = take((size_t)HIDDEN * HIDDEN);
    __bf16* wfi2 = take((size_t)HIDDEN * HIDDEN);
    __bf16* wft2 = take((size_t)HIDDEN * HIDDEN);

    auto CVT = [&](const float* s, __bf16* d, size_t n) {
        long long n4 = (long long)(n / 4);
        int blocks = (int)((n4 + 255) / 256);
        cvt_f32_bf16_kernel<<<blocks, 256, 0, stream>>>(s, d, n4);
    };
    CVT(i_in,  iB,   (size_t)BATCH * IMG);
    CVT(t_in,  tB,   (size_t)BATCH * TAB);
    CVT(fi1_w, wfi1, (size_t)HIDDEN * IMG);
    CVT(ft1_w, wft1, (size_t)HIDDEN * TAB);
    CVT(ci1_w, wci1, (size_t)HIDDEN * HIDDEN);
    CVT(ct1_w, wct1, (size_t)HIDDEN * HIDDEN);
    CVT(aV_wv, wavv, (size_t)HIDDEN * HIDDEN);
    CVT(aV_wo, wavo, (size_t)HIDDEN * HIDDEN);
    CVT(aT_wv, watv, (size_t)HIDDEN * HIDDEN);
    CVT(aT_wo, wato, (size_t)HIDDEN * HIDDEN);
    CVT(fi2_w, wfi2, (size_t)HIDDEN * HIDDEN);
    CVT(ft2_w, wft2, (size_t)HIDDEN * HIDDEN);

    dim3 blk(256);
    dim3 grd(HIDDEN / BN, BATCH / BM);   // (4, 256)

    // relu + bf16-out
    gemm_bf16_wmma_kernel<1,1,0><<<grd, blk, 0, stream>>>(iB, wfi1, fi1_b, nullptr, a0, IMG, 0);
    gemm_bf16_wmma_kernel<1,1,0><<<grd, blk, 0, stream>>>(tB, wft1, ft1_b, nullptr, a1, TAB, 0);

    // V chain: V_ = aV_wo(aV_wv(relu(ci1(i_)))) + i_ ; out[:, :512] = fi2(V_)
    gemm_bf16_wmma_kernel<1,1,0><<<grd, blk, 0, stream>>>(a0, wci1, ci1_b, nullptr, a2, HIDDEN, 0);
    gemm_bf16_wmma_kernel<0,1,0><<<grd, blk, 0, stream>>>(a2, wavv, aV_bv, nullptr, a3, HIDDEN, 0);
    gemm_bf16_wmma_kernel<0,1,1><<<grd, blk, 0, stream>>>(a3, wavo, aV_bo, a0,      a2, HIDDEN, 0);
    gemm_bf16_wmma_kernel<0,0,0><<<grd, blk, 0, stream>>>(a2, wfi2, fi2_b, nullptr, out, HIDDEN, 2 * HIDDEN);

    // T chain: T_ = aT_wo(aT_wv(relu(ct1(t_)))) + t_ ; out[:, 512:] = ft2(T_)
    gemm_bf16_wmma_kernel<1,1,0><<<grd, blk, 0, stream>>>(a1, wct1, ct1_b, nullptr, a3, HIDDEN, 0);
    gemm_bf16_wmma_kernel<0,1,0><<<grd, blk, 0, stream>>>(a3, watv, aT_bv, nullptr, a2, HIDDEN, 0);
    gemm_bf16_wmma_kernel<0,1,1><<<grd, blk, 0, stream>>>(a2, wato, aT_bo, a1,      a3, HIDDEN, 0);
    gemm_bf16_wmma_kernel<0,0,0><<<grd, blk, 0, stream>>>(a3, wft2, ft2_b, nullptr, out + HIDDEN, HIDDEN, 2 * HIDDEN);
}
